// mha_7404523618949
// MI455X (gfx1250) — matrix operand order repaired and checked by reading, not compiled, not run on hardware
//
#include <hip/hip_runtime.h>

// ---------------------------------------------------------------------------
// MHA forward for B=2, T=2048, D=768, H=12 (head dim 64) on gfx1250 (wave32).
// All GEMMs via v_wmma_f32_16x16x32_bf16; weight tiles staged with
// global_load_async_to_lds_b128 (double-buffered, ASYNCcnt-tracked).
// B-fragments are gathered into registers before the WMMA burst so the 4
// matrix ops issue back-to-back on the XDL pipe (single dscnt wait).
// ---------------------------------------------------------------------------

typedef __bf16 bf16_t;
typedef __bf16 v16bf __attribute__((ext_vector_type(16)));
typedef __bf16 v8bf  __attribute__((ext_vector_type(8)));
typedef float  v8f   __attribute__((ext_vector_type(8)));
typedef float  v4f   __attribute__((ext_vector_type(4)));

static constexpr int kB  = 2;
static constexpr int kT  = 2048;
static constexpr int kD  = 768;
static constexpr int kH  = 12;
static constexpr int kND = 64;
static constexpr int kM  = kB * kT;        // 4096 rows
static constexpr int kBH = kB * kH;        // 24

__device__ __forceinline__ v8f wmma_bf16(v16bf a, v16bf b, v8f c) {
  // (neg_a, A, neg_b, B, c_mod, C, reuse_a, reuse_b)
  return __builtin_amdgcn_wmma_f32_16x16x32_bf16(false, a, false, b, (short)0, c,
                                                 false, false);
}

// Async copy of 16B/lane from global (saddr + 32-bit voffset) into LDS.
__device__ __forceinline__ void async_load_b128(unsigned int lds_byte,
                                                const void* sbase,
                                                unsigned int voff_bytes) {
  asm volatile("global_load_async_to_lds_b128 %0, %1, %2"
               :
               : "v"(lds_byte), "v"(voff_bytes), "s"(sbase)
               : "memory");
}

__device__ __forceinline__ void wait_async0() {
  asm volatile("s_wait_asynccnt 0" ::: "memory");
}

// A-fragment (16x32 bf16) built from an fp32 row.  Contiguous k order (not ISA table): lane M = l%16,
// elements e0..15 -> k = b0..b0+15, b0 = (l<16?0:16).
__device__ __forceinline__ v16bf load_a_frag_f32(const float* __restrict__ rowp,
                                                 int kk, int lane) {
  const int b0 = kk + ((lane & 16) ? 16 : 0);
  const v4f c0 = *(const v4f*)(rowp + b0);
  const v4f c1 = *(const v4f*)(rowp + b0 + 4);
  const v4f c2 = *(const v4f*)(rowp + b0 + 8);
  const v4f c3 = *(const v4f*)(rowp + b0 + 12);
  v16bf a;
#pragma unroll
  for (int i = 0; i < 4; ++i) {
    a[i]      = (__bf16)c0[i];
    a[4 + i]  = (__bf16)c1[i];
    a[8 + i]  = (__bf16)c2[i];
    a[12 + i] = (__bf16)c3[i];
  }
  return a;
}

// A-fragment from a bf16 row (two 16B contiguous chunks).
__device__ __forceinline__ v16bf load_a_frag_bf16(const bf16_t* __restrict__ rowp,
                                                  int kk, int lane) {
  const int b0 = kk + ((lane & 16) ? 16 : 0);
  const v8bf lo = *(const v8bf*)(rowp + b0);
  const v8bf hi = *(const v8bf*)(rowp + b0 + 8);
  v16bf a;
#pragma unroll
  for (int i = 0; i < 8; ++i) { a[i] = lo[i]; a[8 + i] = hi[i]; }
  return a;
}

// ---------------------------------------------------------------------------
// Kernel 1: weight transpose + fp32->bf16  (wT[n*768+k] = w[k*768+n])
// ---------------------------------------------------------------------------
__global__ void wconv_kernel(const float* __restrict__ wq,
                             const float* __restrict__ wk,
                             const float* __restrict__ wv,
                             const float* __restrict__ wo,
                             bf16_t* __restrict__ wt) {
  const int z = blockIdx.y;
  const float* w = (z == 0) ? wq : (z == 1) ? wk : (z == 2) ? wv : wo;
  bf16_t* o = wt + (size_t)z * kD * kD;
  const int e = blockIdx.x * blockDim.x + threadIdx.x;
  if (e < kD * kD) {
    const int n = e / kD, k = e % kD;
    o[e] = (__bf16)w[(size_t)k * kD + n];
  }
}

// ---------------------------------------------------------------------------
// Kernel 2: QKV projection + bias + RoPE, emits bf16 in attention layouts:
//   qh,kh : [BH, T, 64]     vT : [BH, 64, T]
// Block 256 (8 waves), tile 128 rows x 64 cols, wave owns 16 rows x 64 cols.
// Weight tile (64n x 32k bf16 = 4KB) async-staged to LDS, double buffered.
// ---------------------------------------------------------------------------
__global__ __launch_bounds__(256) void qkv_kernel(
    const float* __restrict__ xq, const float* __restrict__ xk,
    const float* __restrict__ xv, const bf16_t* __restrict__ wT3,
    const float* __restrict__ bq, const float* __restrict__ bk,
    const float* __restrict__ bv, const float* __restrict__ cosT,
    const float* __restrict__ sinT, bf16_t* __restrict__ qh,
    bf16_t* __restrict__ kh, bf16_t* __restrict__ vT) {
  __shared__ bf16_t wsm[2][64][32] __attribute__((aligned(32)));  // 8KB

  const int mat = blockIdx.z;
  const float*  x    = (mat == 0) ? xq : (mat == 1) ? xk : xv;
  const bf16_t* w    = wT3 + (size_t)mat * kD * kD;
  const float*  bias = (mat == 0) ? bq : (mat == 1) ? bk : bv;

  const int tid  = threadIdx.x;
  const int lane = tid & 31;
  const int wave = tid >> 5;
  const int l16  = lane & 15;
  const int hi   = (lane >> 4) & 1;
  const int m0   = blockIdx.y * 128 + wave * 16;
  const int n0   = blockIdx.x * 64;

  // Cooperative async staging: thread -> (n-row, 16B chunk) of the tile.
  const int wn = tid >> 2;                 // 0..63
  const int wc = tid & 3;                  // 0..3 (8 bf16 each)
  const unsigned int gvoff = (unsigned int)(((n0 + wn) * kD + wc * 8) * 2);
  const unsigned int lds0 =
      (unsigned int)(unsigned long long)&wsm[0][wn][wc * 8];

  v8f acc[4] = {};
  const float* arow = x + (size_t)(m0 + l16) * kD;

  // Preload k-step 0 into buffer 0.
  async_load_b128(lds0, w, gvoff);
  wait_async0();
  __syncthreads();

  int buf = 0;
  for (int kk = 0; kk < kD; kk += 32) {
    const int nxt = kk + 32;
    if (nxt < kD)  // prefetch next tile into the other buffer
      async_load_b128(lds0 + (unsigned int)((buf ^ 1) * 4096), w,
                      gvoff + (unsigned int)(nxt * 2));

    const v16bf a = load_a_frag_f32(arow, kk, lane);
    v16bf bfrag[4];
#pragma unroll
    for (int nt = 0; nt < 4; ++nt)   // 8 ds_load_b128, single wait
      bfrag[nt] = *(const v16bf*)&wsm[buf][nt * 16 + l16][hi ? 16 : 0];
#pragma unroll
    for (int nt = 0; nt < 4; ++nt)   // back-to-back WMMA burst
      acc[nt] = wmma_bf16(a, bfrag[nt], acc[nt]);

    wait_async0();
    __syncthreads();
    buf ^= 1;
  }

#pragma unroll
  for (int nt = 0; nt < 4; ++nt) {
    const int   n  = n0 + nt * 16 + l16;
    const float bb = bias[n];
    const int   h  = n >> 6, d = n & 63, p = d >> 1;
#pragma unroll
    for (int r = 0; r < 8; ++r) {
      const int   m = m0 + r + 8 * hi;
      const int   t = m & (kT - 1);
      const int   b = m >> 11;
      const float v = acc[nt][r] + bb;
      if (mat < 2) {  // q / k : apply RoPE (adjacent cols sit in adjacent lanes)
        const float c   = cosT[t * 32 + p];
        const float s   = sinT[t * 32 + p];
        const float par = __shfl_xor(v, 1, 32);
        const float out = (d & 1) ? (par * s + v * c) : (v * c - par * s);
        bf16_t* dst = (mat == 0) ? qh : kh;
        dst[(((size_t)(b * kH + h)) * kT + t) * kND + d] = (__bf16)out;
      } else {        // v : store transposed so PV B-fragments are contiguous
        vT[(((size_t)(b * kH + h)) * kND + d) * kT + t] = (__bf16)v;
      }
    }
  }
}

// ---------------------------------------------------------------------------
// Kernel 3: causal flash attention.  Block 128 (4 waves); wave = 16 q-rows.
// 32-key steps: 4 WMMA (S), online softmax, P via LDS transpose, 4 WMMA (PV).
// K/V fragments are wave-private -> direct 32B vector loads (L2-resident).
// ---------------------------------------------------------------------------
__global__ __launch_bounds__(128) void attn_kernel(const bf16_t* __restrict__ qh,
                                                   const bf16_t* __restrict__ kh,
                                                   const bf16_t* __restrict__ vT,
                                                   bf16_t* __restrict__ aout) {
  __shared__ bf16_t plds[4][16][32] __attribute__((aligned(32)));  // 4KB

  const int lane = threadIdx.x & 31;
  const int wave = threadIdx.x >> 5;
  const int l16  = lane & 15;
  const int hi   = (lane >> 4) & 1;
  const int bh   = blockIdx.y;
  const int b    = bh / kH, h = bh % kH;
  const int q0   = blockIdx.x * 64 + wave * 16;

  const size_t rowbase = (size_t)bh * kT * kND;
  const bf16_t* qrow   = qh + rowbase + (size_t)(q0 + l16) * kND;
  v16bf qa[2];
  qa[0] = load_a_frag_bf16(qrow, 0, lane);
  qa[1] = load_a_frag_bf16(qrow, 32, lane);

  float mrow[8], lrow[8];
  v8f   o[4] = {};
#pragma unroll
  for (int r = 0; r < 8; ++r) { mrow[r] = -1e30f; lrow[r] = 0.f; }

  const int nkb = (q0 + 47) >> 5;  // causal: only key blocks with kb*32 <= q0+15
  for (int kb = 0; kb < nkb; ++kb) {
    const int k32 = kb * 32;
    v8f s0 = {}, s1 = {};
    {
      const bf16_t* k0 = kh + rowbase + (size_t)(k32 + l16) * kND + (hi ? 16 : 0);
      const bf16_t* k1 = kh + rowbase + (size_t)(k32 + 16 + l16) * kND + (hi ? 16 : 0);
      const v16bf b00 = *(const v16bf*)(k0);
      const v16bf b01 = *(const v16bf*)(k0 + 32);
      const v16bf b10 = *(const v16bf*)(k1);
      const v16bf b11 = *(const v16bf*)(k1 + 32);
      s0 = wmma_bf16(qa[0], b00, s0);
      s0 = wmma_bf16(qa[1], b01, s0);
      s1 = wmma_bf16(qa[0], b10, s1);
      s1 = wmma_bf16(qa[1], b11, s1);
    }

    const bool needMask = (k32 + 31) > q0;
#pragma unroll
    for (int r = 0; r < 8; ++r) {
      s0[r] *= 0.125f;  // 1/sqrt(64)
      s1[r] *= 0.125f;
      if (needMask) {
        const int row = q0 + r + 8 * hi;
        if (k32 + l16 > row)      s0[r] = -1e30f;
        if (k32 + 16 + l16 > row) s1[r] = -1e30f;
      }
    }

    float corr[8];
#pragma unroll
    for (int r = 0; r < 8; ++r) {
      float v = fmaxf(s0[r], s1[r]);
#pragma unroll
      for (int off = 8; off >= 1; off >>= 1) v = fmaxf(v, __shfl_xor(v, off, 32));
      const float mnew = fmaxf(mrow[r], v);
      corr[r] = __expf(mrow[r] - mnew);
      mrow[r] = mnew;
      const float p0 = __expf(s0[r] - mnew);
      const float p1 = __expf(s1[r] - mnew);
      s0[r] = p0; s1[r] = p1;
      float sum = p0 + p1;
#pragma unroll
      for (int off = 8; off >= 1; off >>= 1) sum += __shfl_xor(sum, off, 32);
      lrow[r] = lrow[r] * corr[r] + sum;
#pragma unroll
      for (int dt = 0; dt < 4; ++dt) o[dt][r] *= corr[r];
    }

    // C-layout -> A-layout transpose of P through wave-private LDS.
#pragma unroll
    for (int r = 0; r < 8; ++r) {
      const int row = r + 8 * hi;
      plds[wave][row][l16]      = (__bf16)s0[r];
      plds[wave][row][16 + l16] = (__bf16)s1[r];
    }
    const int  pb  = hi ? 16 : 0;
    const v8bf plo = *(const v8bf*)&plds[wave][l16][pb];
    const v8bf phi = *(const v8bf*)&plds[wave][l16][pb + 8];
    v16bf pa;
#pragma unroll
    for (int i = 0; i < 8; ++i) { pa[i] = plo[i]; pa[8 + i] = phi[i]; }

    // Gather all 4 V fragments first, then a back-to-back WMMA burst.
    const bf16_t* vbase = vT + ((size_t)bh * kND) * kT + k32 + (hi ? 16 : 0);
    v16bf vb[4];
#pragma unroll
    for (int dt = 0; dt < 4; ++dt)
      vb[dt] = *(const v16bf*)(vbase + (size_t)(dt * 16 + l16) * kT);
#pragma unroll
    for (int dt = 0; dt < 4; ++dt) o[dt] = wmma_bf16(pa, vb[dt], o[dt]);
  }

#pragma unroll
  for (int dt = 0; dt < 4; ++dt) {
#pragma unroll
    for (int r = 0; r < 8; ++r) {
      const int   row = q0 + r + 8 * hi;  // token index
      const float val = o[dt][r] / lrow[r];
      aout[((size_t)(b * kT) + row) * kD + h * kND + dt * 16 + l16] = (__bf16)val;
    }
  }
}

// ---------------------------------------------------------------------------
// Kernel 4: output projection (bf16 x bf16 -> fp32), no bias.
// Same async double-buffered weight staging as the QKV GEMM.
// ---------------------------------------------------------------------------
__global__ __launch_bounds__(256) void oproj_kernel(const bf16_t* __restrict__ aout,
                                                    const bf16_t* __restrict__ woT,
                                                    float* __restrict__ out) {
  __shared__ bf16_t wsm[2][64][32] __attribute__((aligned(32)));  // 8KB

  const int tid  = threadIdx.x;
  const int lane = tid & 31;
  const int wave = tid >> 5;
  const int l16  = lane & 15;
  const int hi   = (lane >> 4) & 1;
  const int m0   = blockIdx.y * 128 + wave * 16;
  const int n0   = blockIdx.x * 64;

  const int wn = tid >> 2;
  const int wc = tid & 3;
  const unsigned int gvoff = (unsigned int)(((n0 + wn) * kD + wc * 8) * 2);
  const unsigned int lds0 =
      (unsigned int)(unsigned long long)&wsm[0][wn][wc * 8];

  v8f acc[4] = {};
  const bf16_t* arow = aout + (size_t)(m0 + l16) * kD;

  async_load_b128(lds0, woT, gvoff);
  wait_async0();
  __syncthreads();

  int buf = 0;
  for (int kk = 0; kk < kD; kk += 32) {
    const int nxt = kk + 32;
    if (nxt < kD)
      async_load_b128(lds0 + (unsigned int)((buf ^ 1) * 4096), woT,
                      gvoff + (unsigned int)(nxt * 2));

    const v16bf a = load_a_frag_bf16(arow, kk, lane);
    v16bf bfrag[4];
#pragma unroll
    for (int nt = 0; nt < 4; ++nt)
      bfrag[nt] = *(const v16bf*)&wsm[buf][nt * 16 + l16][hi ? 16 : 0];
#pragma unroll
    for (int nt = 0; nt < 4; ++nt) acc[nt] = wmma_bf16(a, bfrag[nt], acc[nt]);

    wait_async0();
    __syncthreads();
    buf ^= 1;
  }

#pragma unroll
  for (int nt = 0; nt < 4; ++nt) {
    const int n = n0 + nt * 16 + l16;
#pragma unroll
    for (int r = 0; r < 8; ++r) {
      const int m = m0 + r + 8 * hi;
      out[(size_t)m * kD + n] = acc[nt][r];
    }
  }
}

// ---------------------------------------------------------------------------
extern "C" void kernel_launch(void* const* d_in, const int* in_sizes, int n_in,
                              void* d_out, int out_size, void* d_ws,
                              size_t ws_size, hipStream_t stream) {
  const float* q    = (const float*)d_in[0];
  const float* k    = (const float*)d_in[1];
  const float* v    = (const float*)d_in[2];
  // d_in[3] = mask (unused; causal mask computed analytically)
  const float* wq   = (const float*)d_in[4];
  const float* bq   = (const float*)d_in[5];
  const float* wk   = (const float*)d_in[6];
  const float* bk   = (const float*)d_in[7];
  const float* wv   = (const float*)d_in[8];
  const float* bv   = (const float*)d_in[9];
  const float* wo   = (const float*)d_in[10];
  const float* fcos = (const float*)d_in[11];
  const float* fsin = (const float*)d_in[12];
  float* out        = (float*)d_out;

  // Workspace layout (bf16 elements). Total ~28.5 MB — L2 resident.
  bf16_t* ws = (bf16_t*)d_ws;
  const size_t wTsz  = (size_t)4 * kD * kD;          // 2,359,296
  const size_t actsz = (size_t)kM * kD;              // 3,145,728
  bf16_t* wT   = ws;                                  // [4][768(n)][768(k)]
  bf16_t* qh   = ws + wTsz;                           // [BH][T][64]
  bf16_t* kh   = qh + actsz;                          // [BH][T][64]
  bf16_t* vTp  = kh + actsz;                          // [BH][64][T]
  bf16_t* aout = vTp + actsz;                         // [4096][768]

  wconv_kernel<<<dim3((kD * kD + 255) / 256, 4), 256, 0, stream>>>(wq, wk, wv,
                                                                   wo, wT);
  qkv_kernel<<<dim3(kD / 64, kM / 128, 3), 256, 0, stream>>>(
      q, k, v, wT, bq, bk, bv, fcos, fsin, qh, kh, vTp);
  attn_kernel<<<dim3(kT / 64, kBH), 128, 0, stream>>>(qh, kh, vTp, aout);
  oproj_kernel<<<dim3(kD / 64, kM / 128), 256, 0, stream>>>(
      aout, wT + (size_t)3 * kD * kD, out);
}
